// BahdanauAttention1_3788161155177
// MI455X (gfx1250) — matrix operand order repaired and checked by reading, not compiled, not run on hardware
//
#include <hip/hip_runtime.h>
#include <hip/hip_bf16.h>

#define BB 16
#define SS 8192
#define HH 256

// d_out layout: context_tiled [B,H,S] | context [B,H] | attn [B,S]
#define CTXT_OFF 0
#define CTX_OFF  33554432ull
#define ATTN_OFF 33558528ull

typedef __attribute__((ext_vector_type(16))) __bf16 v16bf;
typedef __attribute__((ext_vector_type(8)))  float  v8f;

static __device__ __forceinline__ unsigned short bf_bits(float f) {
    union { float f; unsigned u; } v; v.f = f;
    unsigned r = v.u + 0x7FFFu + ((v.u >> 16) & 1u);   // round-to-nearest-even
    return (unsigned short)(r >> 16);
}
static __device__ __forceinline__ __bf16 f2bf(float f) {
    unsigned short h = bf_bits(f);
    __bf16 r; __builtin_memcpy(&r, &h, 2); return r;
}
static __device__ __forceinline__ float fast_exp(float x) {
    return __builtin_amdgcn_exp2f(x * 1.4426950408889634f);
}
static __device__ __forceinline__ float fast_tanh(float x) {
    float xx = fminf(fmaxf(x, -9.0f), 9.0f);
    float e = __builtin_amdgcn_exp2f(xx * 2.885390081777927f);  // e^(2x)
    return (e - 1.0f) * __builtin_amdgcn_rcpf(e + 1.0f);
}

// ---- P1: pack W2 [H,H] f32 -> bf16 B-fragment layout ------------------------
// frag index: ((n_tile*8 + c)*32 + lane)*16 + j
// lane 0-15: column n = n_tile*16+lane, K = j<8 ? j : j+8        (h = c*32 + K)
// lane 16-31: column n = n_tile*16+lane-16, K = j<8 ? 8+j : 16+j (h = c*32 + K)
__global__ void pack_w2_kernel(const float* __restrict__ W2,
                               unsigned short* __restrict__ bpack) {
    for (int e = blockIdx.x * blockDim.x + threadIdx.x; e < 65536;
         e += gridDim.x * blockDim.x) {
        int j    = e & 15;
        int lane = (e >> 4) & 31;
        int c    = (e >> 9) & 7;
        int nt   = e >> 12;
        int n    = nt * 16 + (lane & 15);
        int h    = c * 32 + ((lane >= 16) ? 8 : 0) + ((j < 8) ? j : j + 8);
        bpack[e] = bf_bits(W2[n * HH + h]);
    }
}

// ---- P2: h1[b,k] = sum_h hidden[b,h] * W1[k,h] ------------------------------
__global__ void h1_kernel(const float* __restrict__ hidden,
                          const float* __restrict__ W1,
                          float* __restrict__ h1) {
    __shared__ float hid[HH];
    int b = blockIdx.x, k = threadIdx.x;
    hid[k] = hidden[b * HH + k];
    __syncthreads();
    const float* wrow = W1 + k * HH;
    float acc = 0.f;
    #pragma unroll 8
    for (int h = 0; h < HH; ++h) acc += wrow[h] * hid[h];
    h1[b * HH + k] = acc;
}

// ---- helpers for A fragment (16-bit A 16x32 layout) -------------------------
static __device__ __forceinline__ v16bf load_a_frag(const float* __restrict__ p) {
    float4 x0 = *(const float4*)(p);
    float4 x1 = *(const float4*)(p + 4);
    float4 y0 = *(const float4*)(p + 16);
    float4 y1 = *(const float4*)(p + 20);
    v16bf v;
    v[0] = f2bf(x0.x);  v[1] = f2bf(x0.y);  v[2]  = f2bf(x0.z);  v[3]  = f2bf(x0.w);
    v[4] = f2bf(x1.x);  v[5] = f2bf(x1.y);  v[6]  = f2bf(x1.z);  v[7]  = f2bf(x1.w);
    v[8] = f2bf(y0.x);  v[9] = f2bf(y0.y);  v[10] = f2bf(y0.z);  v[11] = f2bf(y0.w);
    v[12] = f2bf(y1.x); v[13] = f2bf(y1.y); v[14] = f2bf(y1.z);  v[15] = f2bf(y1.w);
    return v;
}

// Issue one 32 KB stage of packed W2 into LDS via the CDNA5 async copy engine.
// Each of 256 threads moves 8 x 16B; per wave that is 8 async instructions
// (ASYNCcnt-tracked), overlapped with WMMA compute on the other LDS buffer.
static __device__ __forceinline__ void issue_stage_async(
        const unsigned short* __restrict__ bpack, int stage,
        unsigned lds_buf_off, int tid) {
    const unsigned long long gbase =
        (unsigned long long)(const char*)bpack + (unsigned)stage * 32768u +
        (unsigned)tid * 16u;
    const unsigned lbase =
        __builtin_amdgcn_groupstaticsize() + lds_buf_off + (unsigned)tid * 16u;
    #pragma unroll
    for (int k = 0; k < 8; ++k) {
        unsigned lo = lbase + (unsigned)k * 4096u;
        unsigned long long ga = gbase + (unsigned)k * 4096u;
        asm volatile("global_load_async_to_lds_b128 %0, %1, off"
                     :: "v"(lo), "v"(ga) : "memory");
    }
}
static __device__ __forceinline__ void wait_async_done() {
    asm volatile("s_wait_asynccnt 0x0" ::: "memory");
}

// ---- K1: scores[b,s] = sum_k tanh(h1[b,k] + (enc W2^T)[b,s,k]) * V[k] -------
// 8 waves/block, 32 s-rows per wave (two 16-row A tiles share each B fragment).
// Packed W2 (128 KB) streamed through 2 x 32 KB LDS buffers with async copies
// double-buffered against the WMMA loop.
__global__ void __launch_bounds__(256)
scores_kernel(const float* __restrict__ enc,
              const unsigned short* __restrict__ bpack,
              const float* __restrict__ h1g,
              const float* __restrict__ Vg,
              float* __restrict__ scores) {
    extern __shared__ char smem[];                 // 65536 bytes = 2 buffers

    const int tid  = threadIdx.x;
    const int b    = blockIdx.y;
    const int wave = tid >> 5;
    const int lane = tid & 31;
    const int half = lane >> 4;                    // 0: lanes 0-15, 1: lanes 16-31
    const int l16  = lane & 15;
    const int koff = half * 8;
    const int s_base = blockIdx.x * 256 + wave * 32;

    // Kick off stage 0 copy immediately; A-fragment loads overlap it.
    issue_stage_async(bpack, 0, 0u, tid);

    // Preload A fragments: rows s_base+l16 (tile0) and +16 (tile1), all 8 h-chunks
    v16bf a0[8], a1[8];
    const size_t row0 = ((size_t)b * SS + (size_t)(s_base + l16)) * HH;
    const size_t row1 = row0 + (size_t)16 * HH;
    #pragma unroll
    for (int c = 0; c < 8; ++c) {
        a0[c] = load_a_frag(enc + row0 + c * 32 + koff);
        a1[c] = load_a_frag(enc + row1 + c * 32 + koff);
    }

    float sc0[8] = {0, 0, 0, 0, 0, 0, 0, 0};
    float sc1[8] = {0, 0, 0, 0, 0, 0, 0, 0};

    wait_async_done();
    __syncthreads();

    for (int st = 0; st < 4; ++st) {               // 4 stages x 4 n-tiles
        // Prefetch next stage into the other buffer while computing this one.
        if (st < 3) issue_stage_async(bpack, st + 1, (unsigned)((st + 1) & 1) * 32768u, tid);

        const char* bp = smem + (st & 1) * 32768;
        for (int nl = 0; nl < 4; ++nl) {
            const int n = st * 4 + nl;
            v8f acc0 = {}; v8f acc1 = {};
            #pragma unroll
            for (int c = 0; c < 8; ++c) {
                v16bf bf = ((const v16bf*)bp)[(nl * 8 + c) * 32 + lane];
                acc0 = __builtin_amdgcn_wmma_f32_16x16x32_bf16(
                           false, a0[c], false, bf, (short)0, acc0, false, false);
                acc1 = __builtin_amdgcn_wmma_f32_16x16x32_bf16(
                           false, a1[c], false, bf, (short)0, acc1, false, false);
            }
            // lane holds column k = n*16 + l16; VGPR v is row v + 8*half
            const int k = n * 16 + l16;
            const float hv = h1g[(size_t)b * HH + k];
            const float vv = Vg[k];
            #pragma unroll
            for (int v = 0; v < 8; ++v) {
                sc0[v] += fast_tanh(acc0[v] + hv) * vv;
                sc1[v] += fast_tanh(acc1[v] + hv) * vv;
            }
        }

        wait_async_done();                          // next stage fully in LDS
        __syncthreads();                            // all waves done with current buf
    }

    // reduce across the 16 lanes of each half-group
    #pragma unroll
    for (int v = 0; v < 8; ++v) {
        float t0 = sc0[v], t1 = sc1[v];
        for (int m = 1; m <= 8; m <<= 1) {
            t0 += __shfl_xor(t0, m, 32);
            t1 += __shfl_xor(t1, m, 32);
        }
        sc0[v] = t0; sc1[v] = t1;
    }
    float* srow = scores + (size_t)b * SS;
    if (lane == 0) {
        #pragma unroll
        for (int v = 0; v < 8; ++v) {
            srow[s_base + v]      = sc0[v];   // rows 0-7 of tile0
            srow[s_base + 16 + v] = sc1[v];   // rows 0-7 of tile1
        }
    } else if (lane == 16) {
        #pragma unroll
        for (int v = 0; v < 8; ++v) {
            srow[s_base + 8 + v]  = sc0[v];   // rows 8-15 of tile0
            srow[s_base + 24 + v] = sc1[v];   // rows 8-15 of tile1
        }
    }
}

// ---- K2: per-batch softmax stats (max, sum exp) -----------------------------
__global__ void softmax_stats_kernel(const float* __restrict__ scores,
                                     float* __restrict__ mZ) {
    __shared__ float red[256];
    const int b = blockIdx.x, tid = threadIdx.x;
    const float* srow = scores + (size_t)b * SS;
    float m = -1e30f;
    for (int s = tid; s < SS; s += 256) m = fmaxf(m, srow[s]);
    red[tid] = m; __syncthreads();
    for (int o = 128; o; o >>= 1) {
        if (tid < o) red[tid] = fmaxf(red[tid], red[tid + o]);
        __syncthreads();
    }
    const float gm = red[0];
    __syncthreads();
    float z = 0.f;
    for (int s = tid; s < SS; s += 256) z += fast_exp(srow[s] - gm);
    red[tid] = z; __syncthreads();
    for (int o = 128; o; o >>= 1) {
        if (tid < o) red[tid] += red[tid + o];
        __syncthreads();
    }
    if (tid == 0) { mZ[2 * b] = gm; mZ[2 * b + 1] = red[0]; }
}

// ---- K3: attn weights + partial contexts over 128-row chunks ----------------
__global__ void attn_context_kernel(const float* __restrict__ enc,
                                    const float* __restrict__ scores,
                                    const float* __restrict__ mZ,
                                    float* __restrict__ pc,
                                    float* __restrict__ out) {
    __shared__ float w[128];
    const int b = blockIdx.y, chunk = blockIdx.x, tid = threadIdx.x;
    const float gm = mZ[2 * b];
    const float invZ = 1.0f / mZ[2 * b + 1];
    if (tid < 128) {
        const int s = chunk * 128 + tid;
        const float a = fast_exp(scores[(size_t)b * SS + s] - gm) * invZ;
        w[tid] = a;
        out[ATTN_OFF + (size_t)b * SS + s] = a;
    }
    __syncthreads();
    const int h = tid;
    const float* base = enc + ((size_t)b * SS + (size_t)chunk * 128) * HH + h;
    float acc = 0.f;
    #pragma unroll 4
    for (int i = 0; i < 128; ++i) acc += w[i] * base[(size_t)i * HH];
    pc[((size_t)b * 64 + chunk) * HH + h] = acc;
}

// ---- K4: reduce partial contexts -> context [B,H] ---------------------------
__global__ void ctx_reduce_kernel(const float* __restrict__ pc,
                                  float* __restrict__ out) {
    const int b = blockIdx.x, h = threadIdx.x;
    float acc = 0.f;
    #pragma unroll 8
    for (int c = 0; c < 64; ++c) acc += pc[((size_t)b * 64 + c) * HH + h];
    out[CTX_OFF + (size_t)b * HH + h] = acc;
}

// ---- K5: broadcast context -> context_tiled [B,H,S] -------------------------
__global__ void tile_kernel(float* __restrict__ out) {
    const int row = blockIdx.x;                        // b*H + h
    const float v = out[CTX_OFF + row];
    const float4 v4 = make_float4(v, v, v, v);
    float4* dst = (float4*)(out + (size_t)row * SS);
    #pragma unroll 4
    for (int i = threadIdx.x; i < SS / 4; i += 256) dst[i] = v4;
}

extern "C" void kernel_launch(void* const* d_in, const int* in_sizes, int n_in,
                              void* d_out, int out_size, void* d_ws, size_t ws_size,
                              hipStream_t stream) {
    const float* hidden = (const float*)d_in[0];
    const float* enc    = (const float*)d_in[1];
    const float* W1     = (const float*)d_in[2];
    const float* W2     = (const float*)d_in[3];
    const float* V      = (const float*)d_in[4];
    float* out = (float*)d_out;

    char* ws = (char*)d_ws;
    unsigned short* bpack = (unsigned short*)ws;          // 131072 B
    float* h1     = (float*)(ws + 131072);                //  16384 B
    float* scores = (float*)(ws + 147456);                // 524288 B
    float* mZ     = (float*)(ws + 671744);                //    128 B
    float* pc     = (float*)(ws + 671872);                //  1 MiB

    pack_w2_kernel<<<64, 256, 0, stream>>>(W2, bpack);
    h1_kernel<<<BB, 256, 0, stream>>>(hidden, W1, h1);
    scores_kernel<<<dim3(SS / 256, BB), 256, 65536, stream>>>(enc, bpack, h1, V, scores);
    softmax_stats_kernel<<<BB, 256, 0, stream>>>(scores, mZ);
    attn_context_kernel<<<dim3(SS / 128, BB), 256, 0, stream>>>(enc, scores, mZ, pc, out);
    ctx_reduce_kernel<<<BB, 256, 0, stream>>>(pc, out);
    tile_kernel<<<BB * HH, 256, 0, stream>>>(out);
}
